// KK_MultiHeadAttention_11991548690606
// MI455X (gfx1250) — matrix operand order repaired and checked by reading, not compiled, not run on hardware
//
#include <hip/hip_runtime.h>

typedef __bf16 bf16_t;
typedef __attribute__((ext_vector_type(16))) __bf16 v16bf;
typedef __attribute__((ext_vector_type(8)))  float  v8f;
typedef __attribute__((ext_vector_type(4)))  unsigned int u32x4;
typedef __attribute__((ext_vector_type(8)))  int i32x8;
typedef __attribute__((ext_vector_type(4)))  int i32x4;

#if defined(__gfx1250__) && __has_builtin(__builtin_amdgcn_tensor_load_to_lds) && \
    __has_builtin(__builtin_amdgcn_s_wait_tensorcnt)
#define HAVE_TDM 1
#else
#define HAVE_TDM 0
#endif

union FragU { v16bf v; uint4 q[2]; };

__device__ __forceinline__ v8f zero_v8f() {
  v8f z = {0.f,0.f,0.f,0.f,0.f,0.f,0.f,0.f};
  return z;
}

__device__ __forceinline__ v8f wmma_bf16(v16bf a, v16bf b, v8f c) {
  return __builtin_amdgcn_wmma_f32_16x16x32_bf16(false, a, false, b, (short)0, c, false, false);
}

__device__ __forceinline__ bf16_t u16_to_bf16(unsigned short x) {
  union { unsigned short s; bf16_t b; } u; u.s = x; return u.b;
}

// A fragment (16x32 bf16, row m = lane%16): lane needs elements [8*hi,8*hi+8) and
// [16+8*hi, 16+8*hi+8) of its row -> two b128 LDS loads. pitch must be mult of 8.
__device__ __forceinline__ v16bf frag_A(const bf16_t* base, int pitch) {
  int lane = threadIdx.x & 31;
  int m = lane & 15, hi = lane >> 4;
  const bf16_t* r = base + m * pitch + 8 * hi;
  FragU f;
  f.q[0] = *(const uint4*)(r);
  f.q[1] = *(const uint4*)(r + 16);
  return f.v;
}

// B fragment (32x16 bf16 stored transposed [n][k]): lane needs elements
// [8*hi,8*hi+8) and [16+8*hi, 16+8*hi+8) of row n -> two b128 LDS loads.
__device__ __forceinline__ v16bf frag_B(const bf16_t* base, int pitch) {
  int lane = threadIdx.x & 31;
  int n = lane & 15, hi = lane >> 4;
  const bf16_t* r = base + n * pitch + 8 * hi;
  FragU f;
  f.q[0] = *(const uint4*)(r);
  f.q[1] = *(const uint4*)(r + 16);
  return f.v;
}

#if HAVE_TDM
// low 32 bits of a generic pointer to LDS == LDS byte offset (flat->LDS truncation)
__device__ __forceinline__ unsigned lds_addr_of(const void* p) {
  return (unsigned)(unsigned long long)p;
}

// 2-D tile DMA: data_size = 8B units. d0_ctl carries data_size/pad config.
__device__ __forceinline__ void tdm_load_2d(unsigned lds_byte_off, const void* gptr,
                                            unsigned d0_ctl, unsigned t0_units,
                                            unsigned t1_rows, unsigned stride_units) {
  unsigned long long ga = (unsigned long long)gptr;
  u32x4 g0;
  g0[0] = 1u;                                            // count=1, user descriptor
  g0[1] = lds_byte_off;                                  // lds_addr
  g0[2] = (unsigned)ga;                                  // global_addr[31:0]
  g0[3] = ((unsigned)(ga >> 32) & 0x01ffffffu) | 0x80000000u;  // addr[56:32] | type=2
  i32x8 g1;
  g1[0] = (int)d0_ctl;                                   // mask=0 | data_size | pad cfg
  g1[1] = (int)(t0_units << 16);                         // tensor_dim0[15:0]
  g1[2] = (int)((t0_units >> 16) | (t1_rows << 16));     // tensor_dim0 hi | tensor_dim1 lo
  g1[3] = (int)((t1_rows >> 16) | (t0_units << 16));     // tensor_dim1 hi | tile_dim0
  g1[4] = (int)(t1_rows & 0xffffu);                      // tile_dim1 | tile_dim2=0
  g1[5] = (int)stride_units;                             // tensor_dim0_stride[31:0]
  g1[6] = 0;
  g1[7] = 0;
  i32x4 z4 = {0, 0, 0, 0};
  i32x8 z8 = {0, 0, 0, 0, 0, 0, 0, 0};
  __builtin_amdgcn_tensor_load_to_lds(g0, g1, z4, z4, z8, 0);
}
#endif

// ---------------- elementwise conversion kernels ----------------

__global__ void conv_bf16(const float* __restrict__ src, bf16_t* __restrict__ dst, int n) {
  int i = blockIdx.x * 256 + threadIdx.x;
  if (i < n) dst[i] = (bf16_t)src[i];
}

// src f32 [R][C] -> dst bf16 [C][R]
__global__ void convT_bf16(const float* __restrict__ src, bf16_t* __restrict__ dst, int R, int C) {
  int i = blockIdx.x * 256 + threadIdx.x;
  if (i < R * C) {
    int r = i / C, c = i % C;
    dst[(size_t)c * R + r] = (bf16_t)src[i];
  }
}

// ---------------- bf16 GEMM: C[M][N] = A[M][K] * Bt[N][K]^T ----------------
// 256 threads = 8 waves; macro tile 128x128; wave tile 32x64; K step 32.
// LDS pitch 40 elems (80B): matches TDM pad (64B row + 16B pad).

template <bool BF16OUT>
__global__ __launch_bounds__(256) void gemm_kernel(const bf16_t* __restrict__ A,
                                                   const bf16_t* __restrict__ Bt,
                                                   float* __restrict__ Cf,
                                                   bf16_t* __restrict__ Cb,
                                                   int M, int N, int K) {
  __shared__ alignas(16) bf16_t Al[2][128 * 40];
  __shared__ alignas(16) bf16_t Bl[2][128 * 40];
  const int tid = threadIdx.x;
  const int m0 = blockIdx.y * 128, n0 = blockIdx.x * 128;
  const int w = tid >> 5, lane = tid & 31;
  const int wm = w >> 1, wn = w & 1;
  const int hi = lane >> 4, nlo = lane & 15;

  v8f acc[2][4];
#pragma unroll
  for (int mf = 0; mf < 2; ++mf)
#pragma unroll
    for (int nf = 0; nf < 4; ++nf) acc[mf][nf] = zero_v8f();

#if HAVE_TDM
  // data_size=8B (3<<16), pad_enable, pad_interval=16dw (3<<22), pad_amount=4dw (3<<25)
  const unsigned ctl = (3u << 16) | (1u << 20) | (3u << 22) | (3u << 25);
  const unsigned su = (unsigned)(K >> 2);  // row stride in 8B units
  if (w == 0) {
    tdm_load_2d(lds_addr_of(&Al[0][0]), A + (size_t)m0 * K, ctl, 8, 128, su);
    tdm_load_2d(lds_addr_of(&Bl[0][0]), Bt + (size_t)n0 * K, ctl, 8, 128, su);
  }
#endif

  for (int k0 = 0; k0 < K; k0 += 32) {
#if HAVE_TDM
    const int cur = (k0 >> 5) & 1;
    if (w == 0) __builtin_amdgcn_s_wait_tensorcnt(0);
    __syncthreads();  // publishes TDM data; also ends reads of buffer cur^1
    if (w == 0 && k0 + 32 < K) {
      tdm_load_2d(lds_addr_of(&Al[cur ^ 1][0]), A + (size_t)m0 * K + k0 + 32, ctl, 8, 128, su);
      tdm_load_2d(lds_addr_of(&Bl[cur ^ 1][0]), Bt + (size_t)n0 * K + k0 + 32, ctl, 8, 128, su);
    }
#else
    const int cur = 0;
    __syncthreads();
#pragma unroll
    for (int it = 0; it < 4; ++it) {
      int li = tid + 256 * it;
      int row = li >> 3, seg = li & 7;
      *(uint2*)(&Al[0][0] + row * 40 + seg * 4) =
          *(const uint2*)(A + (size_t)(m0 + row) * K + k0 + seg * 4);
      *(uint2*)(&Bl[0][0] + row * 40 + seg * 4) =
          *(const uint2*)(Bt + (size_t)(n0 + row) * K + k0 + seg * 4);
    }
    if (k0 + 32 < K) {
      __builtin_prefetch(A + (size_t)(m0 + (tid >> 1)) * K + k0 + 32, 0, 1);
      __builtin_prefetch(Bt + (size_t)(n0 + (tid >> 1)) * K + k0 + 32, 0, 1);
    }
    __syncthreads();
#endif

    v16bf af[2], bfv[4];
#pragma unroll
    for (int mf = 0; mf < 2; ++mf) af[mf] = frag_A(&Al[cur][0] + (wm * 32 + mf * 16) * 40, 40);
#pragma unroll
    for (int nf = 0; nf < 4; ++nf) bfv[nf] = frag_B(&Bl[cur][0] + (wn * 64 + nf * 16) * 40, 40);
#pragma unroll
    for (int mf = 0; mf < 2; ++mf)
#pragma unroll
      for (int nf = 0; nf < 4; ++nf) acc[mf][nf] = wmma_bf16(af[mf], bfv[nf], acc[mf][nf]);
  }

#pragma unroll
  for (int mf = 0; mf < 2; ++mf)
#pragma unroll
    for (int nf = 0; nf < 4; ++nf)
#pragma unroll
      for (int e = 0; e < 8; ++e) {
        int m = m0 + wm * 32 + mf * 16 + e + 8 * hi;
        int n = n0 + wn * 64 + nf * 16 + nlo;
        if (BF16OUT) Cb[(size_t)m * N + n] = (bf16_t)acc[mf][nf][e];
        else         Cf[(size_t)m * N + n] = acc[mf][nf][e];
      }
}

// ---------------- fused rel-shift flash attention ----------------
// grid (8, BS=8, H=16), block 128 (4 waves). Each WG: 64 query rows of one (b,h);
// each wave: 16 rows, online softmax over j in chunks of 32.
// rel_shift + causal mask reduce to: pos[i][j] = qv[i] . p[511 - i + j], j <= i+512.

__global__ __launch_bounds__(128) void attn_kernel(const float* __restrict__ Qf,
                                                   const bf16_t* __restrict__ KV,
                                                   const bf16_t* __restrict__ Pp,
                                                   const float* __restrict__ u,
                                                   const float* __restrict__ v,
                                                   bf16_t* __restrict__ AWV) {
  const int i0 = blockIdx.x * 64;
  const int b  = blockIdx.y;
  const int h  = blockIdx.z;
  const int tid = threadIdx.x, lane = tid & 31, w = tid >> 5;
  const int half = lane >> 4, rloc = lane & 15;
  const int iw0 = i0 + 16 * w;
  const float NEG_INF = -__builtin_inff();

  __shared__ alignas(16) bf16_t Qu[64 * 64];
  __shared__ alignas(16) bf16_t Qv[64 * 64];
  __shared__ alignas(16) bf16_t Kl[32 * 64];
  __shared__ alignas(16) bf16_t Vt[64 * 32];
  __shared__ alignas(16) bf16_t Pl[96 * 64];
  __shared__ alignas(16) float  SC[4][16 * 32];
  __shared__ alignas(16) float  SP[4][16 * 48];
  __shared__ alignas(16) bf16_t Pb[4][16 * 32];
  __shared__ float  mrow[4][16], lrow[4][16], crow[4][16];

  // stage Q block, add u/v, convert to bf16
#pragma unroll 4
  for (int it = 0; it < 32; ++it) {
    int e = tid + 128 * it;
    int r = e >> 6, d = e & 63;
    float q = Qf[(size_t)((i0 + r) * 8 + b) * 1024 + h * 64 + d];
    Qu[r * 64 + d] = (bf16_t)(q + u[h * 64 + d]);
    Qv[r * 64 + d] = (bf16_t)(q + v[h * 64 + d]);
  }
  if (lane < 16) {
    mrow[w][lane] = NEG_INF; lrow[w][lane] = 0.f; crow[w][lane] = 0.f;
  }
  v8f accO[4];
#pragma unroll
  for (int dt = 0; dt < 4; ++dt) accO[dt] = zero_v8f();
  __syncthreads();

  const int jmax = i0 + 63 + 512;  // <= 1023 always

  for (int j0 = 0; j0 <= jmax; j0 += 32) {
    const int rbase = 448 - i0 + j0;  // P window [rbase, rbase+96)
#if HAVE_TDM
    if (w == 0) {
      // K rows [j0,j0+32): 128B rows, global row stride 8*2048*2B = 4096 units
      tdm_load_2d(lds_addr_of(Kl), KV + (size_t)(j0 * 8 + b) * 2048 + h * 64,
                  3u << 16, 16, 32, 4096);
      // P rows [rbase, rbase+96): 128B rows, stride 2048B = 256 units
      tdm_load_2d(lds_addr_of(Pl), Pp + (size_t)rbase * 1024 + h * 64,
                  3u << 16, 16, 96, 256);
    }
#else
#pragma unroll
    for (int it = 0; it < 8; ++it) {
      int qd = tid + 128 * it;  // 1024 dwords
      int jj = qd >> 5, seg = qd & 31;
      ((unsigned int*)Kl)[jj * 32 + seg] =
          ((const unsigned int*)(KV + (size_t)((j0 + jj) * 8 + b) * 2048 + h * 64))[seg];
    }
#pragma unroll
    for (int it = 0; it < 24; ++it) {
      int qd = tid + 128 * it;  // 3072 dwords
      int pr = qd >> 5, seg = qd & 31;
      ((unsigned int*)Pl)[pr * 32 + seg] =
          ((const unsigned int*)(Pp + (size_t)(rbase + pr) * 1024 + h * 64))[seg];
    }
#endif
    // V rows transposed into Vt[d][jj] (dword global loads, 2 b16 LDS stores)
#pragma unroll
    for (int it = 0; it < 8; ++it) {
      int e = tid + 128 * it;  // 1024 pairs
      int jj = e >> 5, d2 = (e & 31) * 2;
      unsigned pv = *(const unsigned*)(KV + (size_t)((j0 + jj) * 8 + b) * 2048 + 1024 + h * 64 + d2);
      Vt[d2 * 32 + jj]       = u16_to_bf16((unsigned short)(pv & 0xffffu));
      Vt[(d2 + 1) * 32 + jj] = u16_to_bf16((unsigned short)(pv >> 16));
    }
#if HAVE_TDM
    if (w == 0) __builtin_amdgcn_s_wait_tensorcnt(0);
#endif
    __syncthreads();

    // --- content scores: Qu(16x64) x K^T -> SC 16x32 ---
#pragma unroll
    for (int jt = 0; jt < 2; ++jt) {
      v8f s = zero_v8f();
#pragma unroll
      for (int kk = 0; kk < 2; ++kk)
        s = wmma_bf16(frag_A(Qu + (w * 16) * 64 + kk * 32, 64),
                      frag_B(Kl + (jt * 16) * 64 + kk * 32, 64), s);
#pragma unroll
      for (int e = 0; e < 8; ++e)
        SC[w][(e + 8 * half) * 32 + jt * 16 + rloc] = s[e];
    }
    // --- position scores: Qv x P^T over 3 r-tiles -> SP 16x48 ---
#pragma unroll
    for (int rt = 0; rt < 3; ++rt) {
      v8f s = zero_v8f();
#pragma unroll
      for (int kk = 0; kk < 2; ++kk)
        s = wmma_bf16(frag_A(Qv + (w * 16) * 64 + kk * 32, 64),
                      frag_B(Pl + (48 - 16 * w + rt * 16) * 64 + kk * 32, 64), s);
#pragma unroll
      for (int e = 0; e < 8; ++e)
        SP[w][(e + 8 * half) * 48 + rt * 16 + rloc] = s[e];
    }
    __syncthreads();

    // --- combine + shift + mask + online softmax (2 lanes per row) ---
    {
      int row = rloc;
      int ig = iw0 + row;
      float mo = mrow[w][row];
      float sv[16];
      float smax = NEG_INF;
#pragma unroll
      for (int t = 0; t < 16; ++t) {
        int dj = half * 16 + t;
        int j = j0 + dj;
        float s;
        if (j > ig + 512) s = NEG_INF;  // causal mask (j >= i + 513)
        else s = (SC[w][row * 32 + dj] + SP[w][row * 48 + 15 - row + dj]) * 0.125f;
        sv[t] = s;
        smax = fmaxf(smax, s);
      }
      smax = fmaxf(smax, __shfl_xor(smax, 16));
      float mn = fmaxf(mo, smax);
      float c = __expf(mo - mn);
      float ssum = 0.f;
#pragma unroll
      for (int t = 0; t < 16; ++t) {
        int dj = half * 16 + t;
        float p = __expf(sv[t] - mn);
        ssum += p;
        Pb[w][row * 32 + dj] = (bf16_t)p;
      }
      ssum += __shfl_xor(ssum, 16);
      if (half == 0) {
        mrow[w][row] = mn;
        crow[w][row] = c;
        lrow[w][row] = lrow[w][row] * c + ssum;
      }
    }
    __syncthreads();

    // --- rescale O and accumulate P x V ---
#pragma unroll
    for (int dt = 0; dt < 4; ++dt)
#pragma unroll
      for (int e = 0; e < 8; ++e) accO[dt][e] *= crow[w][e + 8 * half];
    {
      v16bf ap = frag_A(&Pb[w][0], 32);
#pragma unroll
      for (int dt = 0; dt < 4; ++dt)
        accO[dt] = wmma_bf16(ap, frag_B(Vt + (dt * 16) * 32, 32), accO[dt]);
    }
    __syncthreads();
  }

  // --- normalize and store ---
#pragma unroll
  for (int dt = 0; dt < 4; ++dt)
#pragma unroll
    for (int e = 0; e < 8; ++e) {
      int m = e + 8 * half;
      int i = iw0 + m;
      float o = accO[dt][e] / lrow[w][m];
      AWV[(size_t)(i * 8 + b) * 1024 + h * 64 + dt * 16 + rloc] = (bf16_t)o;
    }
}

// ---------------- residual + LayerNorm ----------------

__global__ __launch_bounds__(256) void ln_kernel(const float* __restrict__ inp,
                                                 const float* __restrict__ Of,
                                                 const float* __restrict__ gamma,
                                                 const float* __restrict__ beta,
                                                 float* __restrict__ out) {
  __shared__ float red[256];
  int row = blockIdx.x, tid = threadIdx.x;
  float x[4];
  float s1 = 0.f;
#pragma unroll
  for (int t = 0; t < 4; ++t) {
    int c = tid + 256 * t;
    x[t] = inp[(size_t)row * 1024 + c] + Of[(size_t)row * 1024 + c];
    s1 += x[t];
  }
  red[tid] = s1;
  for (int off = 128; off > 0; off >>= 1) {
    __syncthreads();
    if (tid < off) red[tid] += red[tid + off];
  }
  __syncthreads();
  float mu = red[0] * (1.f / 1024.f);
  __syncthreads();
  float s2 = 0.f;
#pragma unroll
  for (int t = 0; t < 4; ++t) {
    float d = x[t] - mu;
    s2 += d * d;
  }
  red[tid] = s2;
  for (int off = 128; off > 0; off >>= 1) {
    __syncthreads();
    if (tid < off) red[tid] += red[tid + off];
  }
  __syncthreads();
  float var = red[0] * (1.f / 1024.f);
  float rstd = rsqrtf(var + 1e-5f);
#pragma unroll
  for (int t = 0; t < 4; ++t) {
    int c = tid + 256 * t;
    out[(size_t)row * 1024 + c] = (x[t] - mu) * rstd * gamma[c] + beta[c];
  }
}

// ---------------- host launcher ----------------

extern "C" void kernel_launch(void* const* d_in, const int* in_sizes, int n_in,
                              void* d_out, int out_size, void* d_ws, size_t ws_size,
                              hipStream_t stream) {
  const float* input_ = (const float*)d_in[0];
  const float* pos    = (const float*)d_in[1];
  const float* memory = (const float*)d_in[2];
  const float* u      = (const float*)d_in[3];
  const float* v      = (const float*)d_in[4];
  // d_in[5] = mask (computed analytically in-kernel)
  const float* W_kv   = (const float*)d_in[6];
  const float* W_q    = (const float*)d_in[7];
  const float* W_p    = (const float*)d_in[8];
  const float* W_out  = (const float*)d_in[9];
  const float* gamma  = (const float*)d_in[10];
  const float* beta   = (const float*)d_in[11];
  float* out = (float*)d_out;

  char* p = (char*)d_ws;
  auto alloc = [&](size_t bytes) -> char* {
    char* r = p;
    p += (bytes + 255) & ~(size_t)255;
    return r;
  };
  bf16_t* Xall = (bf16_t*)alloc(8192ull * 1024 * 2);  // concat(memory, input) bf16
  bf16_t* Wkvt = (bf16_t*)alloc(2048ull * 1024 * 2);
  bf16_t* Wqt  = (bf16_t*)alloc(1024ull * 1024 * 2);
  bf16_t* Wpt  = (bf16_t*)alloc(1024ull * 1024 * 2);
  bf16_t* Wot  = (bf16_t*)alloc(1024ull * 1024 * 2);
  bf16_t* Pemb = (bf16_t*)alloc(1024ull * 1024 * 2);
  bf16_t* KVb  = (bf16_t*)alloc(8192ull * 2048 * 2);
  float*  Qfb  = (float*)alloc(4096ull * 1024 * 4);
  bf16_t* Ppb  = (bf16_t*)alloc(1088ull * 1024 * 2);  // 64 pad rows (masked band)
  bf16_t* AWV  = (bf16_t*)alloc(4096ull * 1024 * 2);
  float*  Of   = (float*)alloc(4096ull * 1024 * 4);

  const int NMEM = 512 * 8 * 1024;  // 4194304
  conv_bf16<<<(NMEM + 255) / 256, 256, 0, stream>>>(memory, Xall, NMEM);
  conv_bf16<<<(NMEM + 255) / 256, 256, 0, stream>>>(input_, Xall + NMEM, NMEM);
  conv_bf16<<<(1048576 + 255) / 256, 256, 0, stream>>>(pos, Pemb, 1048576);
  convT_bf16<<<(2097152 + 255) / 256, 256, 0, stream>>>(W_kv, Wkvt, 1024, 2048);
  convT_bf16<<<(1048576 + 255) / 256, 256, 0, stream>>>(W_q, Wqt, 1024, 1024);
  convT_bf16<<<(1048576 + 255) / 256, 256, 0, stream>>>(W_p, Wpt, 1024, 1024);
  convT_bf16<<<(1048576 + 255) / 256, 256, 0, stream>>>(W_out, Wot, 1024, 1024);

  gemm_kernel<true><<<dim3(16, 64), 256, 0, stream>>>(Xall, Wkvt, nullptr, KVb, 8192, 2048, 1024);
  gemm_kernel<false><<<dim3(8, 32), 256, 0, stream>>>(Xall + NMEM, Wqt, Qfb, nullptr, 4096, 1024, 1024);
  gemm_kernel<true><<<dim3(8, 8), 256, 0, stream>>>(Pemb, Wpt, nullptr, Ppb, 1024, 1024, 1024);

  attn_kernel<<<dim3(8, 8, 16), 128, 0, stream>>>(Qfb, KVb, Ppb, u, v, AWV);

  gemm_kernel<false><<<dim3(8, 32), 256, 0, stream>>>(AWV, Wot, Of, nullptr, 4096, 1024, 1024);
  ln_kernel<<<4096, 256, 0, stream>>>(input_, Of, gamma, beta, out);

  (void)in_sizes; (void)n_in; (void)out_size; (void)ws_size;
}
